// NanoGPT_91182155694213
// MI455X (gfx1250) — matrix operand order repaired and checked by reading, not compiled, not run on hardware
//
#include <hip/hip_runtime.h>
#include <hip/hip_bf16.h>

// ---------------------------------------------------------------------------
// NanoGPT forward pass for gfx1250 (MI455X).
// Compute-bound (AI ~500-1000 FLOP/B vs 23.3 TB/s HBM) => every GEMM runs on
// the matrix pipe: v_wmma_f32_16x16x32_bf16 (bf16 in, f32 accumulate).
// Data movement: async global->LDS tile staging (ASYNCcnt), double-buffered,
// fragments fed from LDS via ds_load_b128. Falls back to direct global loads
// if the async builtin is unavailable in this toolchain.
// ---------------------------------------------------------------------------

typedef __attribute__((ext_vector_type(16))) __bf16 v16bf;
typedef __attribute__((ext_vector_type(8)))  __bf16 v8bf;
typedef __attribute__((ext_vector_type(8)))  float  v8f;

constexpr int cB  = 4;
constexpr int cT  = 1024;
constexpr int cE  = 1024;
constexpr int cH  = 16;
constexpr int cHD = 64;
constexpr int cNL = 6;
constexpr int cV  = 50257;
constexpr int cBT = cB * cT;          // 4096 rows of activations

#if defined(__AMDGCN__) && __has_builtin(__builtin_amdgcn_global_load_async_to_lds_b128)
#define USE_ASYNC 1
#else
#define USE_ASYNC 0
#endif

#if USE_ASYNC
#define AS1 __attribute__((address_space(1)))
#define AS3 __attribute__((address_space(3)))
// Builtin parameter pointee type per hipcc diagnostic:
//   '__attribute__((__vector_size__(4 * sizeof(int)))) int __device__ *'
typedef int v4i_b __attribute__((vector_size(16)));

__device__ __forceinline__ void wait_asynccnt0() {
#if __has_builtin(__builtin_amdgcn_s_wait_asynccnt)
  __builtin_amdgcn_s_wait_asynccnt(0);
#else
  asm volatile("s_wait_asynccnt 0x0" ::: "memory");
#endif
}
__device__ __forceinline__ void wait_asynccnt4() {
#if __has_builtin(__builtin_amdgcn_s_wait_asynccnt)
  __builtin_amdgcn_s_wait_asynccnt(4);
#else
  asm volatile("s_wait_asynccnt 0x4" ::: "memory");
#endif
}
#endif

// ---- f32 -> bf16 round-to-nearest-even (bit manipulation, no intrinsics) ---
__device__ __forceinline__ __bf16 f2bf(float f) {
  union { float f; unsigned u; } in; in.f = f;
  unsigned u = in.u + 0x7FFFu + ((in.u >> 16) & 1u);
  union { unsigned short s; __bf16 b; } out; out.s = (unsigned short)(u >> 16);
  return out.b;
}

// ---------------------------------------------------------------------------
// Batched WMMA GEMM:  C[y][z] = alpha * A[y][z] * B[y][z]^T (+ C if ACC_)
//   A: M x K row-major (lda), bf16.  B: N x K row-major (ldb), bf16.
//   C: M x N row-major (ldc), f32.
// Batch offsets applied per grid.y / grid.z (covers (b) and (b,h) batching).
// 256 threads = 8 waves per block; block computes a 128x128 C tile; each wave
// owns 64x32 (4 M-tiles x 2 N-tiles) => 8 v_wmma per 32-deep K step.
// With USE_ASYNC: A/B 128x32 tiles are DMA'd into LDS (double-buffered) with
// global_load_async_to_lds_b128; fragments come from ds_load_b128.
// ---------------------------------------------------------------------------
template<bool ACC_>
__global__ __launch_bounds__(256) void gemm_wmma(
    const __bf16* __restrict__ Ag, const __bf16* __restrict__ Bg,
    float* __restrict__ Cg,
    int M, int N, int K, int lda, int ldb, int ldc,
    long sAy, long sBy, long sCy, long sAz, long sBz, long sCz,
    float alpha, int tilesN)
{
  const __bf16* A  = Ag + (long)blockIdx.y * sAy + (long)blockIdx.z * sAz;
  const __bf16* Bm = Bg + (long)blockIdx.y * sBy + (long)blockIdx.z * sBz;
  float* C = Cg + (long)blockIdx.y * sCy + (long)blockIdx.z * sCz;

  const int bm = blockIdx.x / tilesN;
  const int bn = blockIdx.x % tilesN;
  const int wave = threadIdx.x >> 5;
  const int lane = threadIdx.x & 31;
  const int wr = wave >> 2;        // 0..1 (64-row halves)
  const int wc = wave & 3;         // 0..3 (32-col quarters)
  const int rowBase0 = bm * 128;
  const int colBase0 = bn * 128;
  const int rowBase = rowBase0 + wr * 64;
  const int colBase = colBase0 + wc * 32;
  const int lm = lane & 15;
  const int hi = lane >> 4;        // lane half (ISA K-split)
  const int akoff = hi * 8;        // A frag: K {akoff..+7, akoff+16..+23}
  const int bkoff = hi * 8;        // B frag: K {bkoff..+7, bkoff+16..+23}

  v8f zero = {};
  v8f acc[4][2];
#pragma unroll
  for (int i = 0; i < 4; ++i)
#pragma unroll
    for (int j = 0; j < 2; ++j) acc[i][j] = zero;

#if USE_ASYNC
  // ---- async LDS-staged, double-buffered main loop ----
  __shared__ __align__(16) __bf16 tA[2][128 * 32];   // 8 KB per buffer
  __shared__ __align__(16) __bf16 tB[2][128 * 32];   // 8 KB per buffer
  const int tid = threadIdx.x;

  auto stage = [&](int buf, int k0) {
    // A tile: 128 rows x 32 cols bf16 = 512 x 16B chunks; 2 per thread.
#pragma unroll
    for (int c = 0; c < 2; ++c) {
      int q = tid + c * 256;
      int row = q >> 2, o16 = q & 3;               // 4 chunks per row
      int gr = rowBase0 + row; if (gr >= M) gr = M - 1;
      const __bf16* gp = A + (long)gr * lda + k0 + o16 * 8;
      __builtin_amdgcn_global_load_async_to_lds_b128(
          (AS1 v4i_b*)(unsigned long long)gp,
          (AS3 v4i_b*)&tA[buf][q * 8], 0, 0);
    }
    // B tile: 128 n-rows x 32 k-cols.
#pragma unroll
    for (int c = 0; c < 2; ++c) {
      int q = tid + c * 256;
      int row = q >> 2, o16 = q & 3;
      int gc = colBase0 + row; if (gc >= N) gc = N - 1;
      const __bf16* gp = Bm + (long)gc * ldb + k0 + o16 * 8;
      __builtin_amdgcn_global_load_async_to_lds_b128(
          (AS1 v4i_b*)(unsigned long long)gp,
          (AS3 v4i_b*)&tB[buf][q * 8], 0, 0);
    }
  };

  const int nsteps = K >> 5;
  stage(0, 0);
  for (int s = 0; s < nsteps; ++s) {
    const int buf = s & 1;
    const bool pf = (s + 1 < nsteps);
    if (pf) stage(buf ^ 1, (s + 1) << 5);
    if (pf) wait_asynccnt4(); else wait_asynccnt0();  // own wave's stage done
    __syncthreads();                                  // union across waves

    v16bf a[4], b[2];
#pragma unroll
    for (int i = 0; i < 4; ++i) {
      const int rl = wr * 64 + i * 16 + lm;
      const v8bf lo = *(const v8bf*)&tA[buf][rl * 32 + akoff];
      const v8bf hh = *(const v8bf*)&tA[buf][rl * 32 + akoff + 16];
#pragma unroll
      for (int j = 0; j < 8; ++j) { a[i][j] = lo[j]; a[i][8 + j] = hh[j]; }
    }
#pragma unroll
    for (int i = 0; i < 2; ++i) {
      const int nl = wc * 32 + i * 16 + lm;
      const v8bf lo = *(const v8bf*)&tB[buf][nl * 32 + bkoff];
      const v8bf hh = *(const v8bf*)&tB[buf][nl * 32 + bkoff + 16];
#pragma unroll
      for (int j = 0; j < 8; ++j) { b[i][j] = lo[j]; b[i][8 + j] = hh[j]; }
    }
#pragma unroll
    for (int i = 0; i < 4; ++i)
#pragma unroll
      for (int j = 0; j < 2; ++j)
        acc[i][j] = __builtin_amdgcn_wmma_f32_16x16x32_bf16(
            false, a[i], false, b[j], (short)0, acc[i][j], false, false);

    __syncthreads();   // reads done before buffer is overwritten next stage
  }
#else
  // ---- fallback: direct global fragment loads (proven codegen) ----
  for (int k0 = 0; k0 < K; k0 += 32) {
    v16bf a[4], b[2];
#pragma unroll
    for (int i = 0; i < 4; ++i) {
      int r = rowBase + i * 16 + lm; if (r >= M) r = M - 1;
      const __bf16* p = A + (long)r * lda + k0 + akoff;
#pragma unroll
      for (int j = 0; j < 8; ++j) { a[i][j] = p[j]; a[i][8 + j] = p[16 + j]; }
    }
#pragma unroll
    for (int i = 0; i < 2; ++i) {
      int c = colBase + i * 16 + lm; if (c >= N) c = N - 1;
      const __bf16* p = Bm + (long)c * ldb + k0 + bkoff;
#pragma unroll
      for (int j = 0; j < 8; ++j) { b[i][j] = p[j]; b[i][8 + j] = p[16 + j]; }
    }
#pragma unroll
    for (int i = 0; i < 4; ++i)
#pragma unroll
      for (int j = 0; j < 2; ++j)
        acc[i][j] = __builtin_amdgcn_wmma_f32_16x16x32_bf16(
            false, a[i], false, b[j], (short)0, acc[i][j], false, false);
  }
#endif

  // C/D layout: VGPR v -> M = v + 8*hi, N = lane&15.
  const int mOff = hi * 8;
#pragma unroll
  for (int i = 0; i < 4; ++i) {
#pragma unroll
    for (int j = 0; j < 2; ++j) {
      const int col = colBase + j * 16 + lm;
      if (col >= N) continue;
#pragma unroll
      for (int v = 0; v < 8; ++v) {
        const int row = rowBase + i * 16 + mOff + v;
        if (row < M) {
          long cidx = (long)row * ldc + col;
          float val = alpha * acc[i][j][v];
          if (ACC_) C[cidx] = C[cidx] + val; else C[cidx] = val;
        }
      }
    }
  }
}

// ------------------------------ embedding ----------------------------------
__global__ __launch_bounds__(256) void embed_kernel(
    const int* __restrict__ idx, const float* __restrict__ wte,
    const float* __restrict__ wpe, float* __restrict__ x)
{
  long i = (long)blockIdx.x * 256 + threadIdx.x;
  if (i >= (long)cBT * cE) return;
  int e  = (int)(i % cE);
  int bt = (int)(i / cE);
  int t  = bt % cT;
  x[i] = wte[(long)idx[bt] * cE + e] + wpe[(long)t * cE + e];
}

// ------------------------------ layernorm ----------------------------------
__global__ __launch_bounds__(256) void layernorm_bf16_kernel(
    const float* __restrict__ x, const float* __restrict__ w,
    const float* __restrict__ b, __bf16* __restrict__ out)
{
  __shared__ float red[256];
  const int row = blockIdx.x;
  const int tid = threadIdx.x;
  const float* xr = x + (long)row * cE;

  float s = 0.f;
  for (int i = tid; i < cE; i += 256) s += xr[i];
  red[tid] = s; __syncthreads();
  for (int off = 128; off > 0; off >>= 1) {
    if (tid < off) red[tid] += red[tid + off];
    __syncthreads();
  }
  const float mu = red[0] * (1.0f / cE);
  __syncthreads();

  float v = 0.f;
  for (int i = tid; i < cE; i += 256) { float d = xr[i] - mu; v += d * d; }
  red[tid] = v; __syncthreads();
  for (int off = 128; off > 0; off >>= 1) {
    if (tid < off) red[tid] += red[tid + off];
    __syncthreads();
  }
  const float rstd = rsqrtf(red[0] * (1.0f / cE) + 1e-5f);

  __bf16* orow = out + (long)row * cE;
  for (int i = tid; i < cE; i += 256)
    orow[i] = f2bf((xr[i] - mu) * rstd * w[i] + b[i]);
}

// -------------------------- causal softmax ---------------------------------
__global__ __launch_bounds__(256) void attn_softmax_kernel(float* __restrict__ att)
{
  __shared__ float red[256];
  const long r = blockIdx.x;          // row in (B,H,T)
  const int  q = (int)(r & (cT - 1)); // query position -> attend k<=q
  float* row = att + r * (long)cT;
  const int tid = threadIdx.x;

  float mx = -3.4e38f;
  for (int i = tid; i <= q; i += 256) mx = fmaxf(mx, row[i]);
  red[tid] = mx; __syncthreads();
  for (int off = 128; off > 0; off >>= 1) {
    if (tid < off) red[tid] = fmaxf(red[tid], red[tid + off]);
    __syncthreads();
  }
  mx = red[0]; __syncthreads();

  float s = 0.f;
  for (int i = tid; i <= q; i += 256) {
    float e = __expf(row[i] - mx);
    row[i] = e; s += e;
  }
  red[tid] = s; __syncthreads();
  for (int off = 128; off > 0; off >>= 1) {
    if (tid < off) red[tid] += red[tid + off];
    __syncthreads();
  }
  const float inv = 1.0f / red[0];
  for (int i = tid; i < cT; i += 256)
    row[i] = (i <= q) ? row[i] * inv : 0.0f;
}

// ---------------------- V transpose (per b,h: T x hd -> hd x T) ------------
__global__ __launch_bounds__(256) void transpose_v_kernel(
    const __bf16* __restrict__ qkvb, __bf16* __restrict__ vt)
{
  long i = (long)blockIdx.x * 256 + threadIdx.x;   // over B*H*HD*T
  if (i >= (long)cB * cH * cHD * cT) return;
  int t = (int)(i % cT);
  long r = i / cT;             // (b*H + h)*HD + d
  int d = (int)(r % cHD);
  long bh = r / cHD;
  int h = (int)(bh % cH);
  int b = (int)(bh / cH);
  vt[i] = qkvb[((long)b * cT + t) * (3 * cE) + 2 * cE + (long)h * cHD + d];
}

// --------------------------- elementwise -----------------------------------
__global__ __launch_bounds__(256) void cvt_bf16_kernel(
    const float* __restrict__ in, __bf16* __restrict__ out, long n)
{
  long i = (long)blockIdx.x * 256 + threadIdx.x;
  if (i < n) out[i] = f2bf(in[i]);
}

__global__ __launch_bounds__(256) void gelu_bf16_kernel(
    const float* __restrict__ in, __bf16* __restrict__ out, long n)
{
  long i = (long)blockIdx.x * 256 + threadIdx.x;
  if (i < n) {
    float v = in[i];
    out[i] = f2bf(0.5f * v * (1.0f + erff(v * 0.70710678118654752f)));
  }
}

// --------------------------- cross entropy ---------------------------------
__global__ void zero_loss_kernel(float* acc) { acc[0] = 0.f; acc[1] = 0.f; }

__global__ __launch_bounds__(256) void ce_kernel(
    const float* __restrict__ logits, const int* __restrict__ tgt,
    float* __restrict__ acc)
{
  __shared__ float red[256];
  const int r = blockIdx.x;
  const int tid = threadIdx.x;
  const float* lr = logits + (long)r * cV;

  float mx = -3.4e38f;
  for (int i = tid; i < cV; i += 256) mx = fmaxf(mx, lr[i]);
  red[tid] = mx; __syncthreads();
  for (int off = 128; off > 0; off >>= 1) {
    if (tid < off) red[tid] = fmaxf(red[tid], red[tid + off]);
    __syncthreads();
  }
  mx = red[0]; __syncthreads();

  float s = 0.f;
  for (int i = tid; i < cV; i += 256) s += __expf(lr[i] - mx);
  red[tid] = s; __syncthreads();
  for (int off = 128; off > 0; off >>= 1) {
    if (tid < off) red[tid] += red[tid + off];
    __syncthreads();
  }
  if (tid == 0) {
    int t = tgt[r];
    if (t >= 0) {
      float nll = logf(red[0]) + mx - lr[t];
      atomicAdd(&acc[0], nll);
      atomicAdd(&acc[1], 1.0f);
    }
  }
}

__global__ void ce_finalize_kernel(const float* acc, float* out)
{
  out[0] = acc[0] / fmaxf(acc[1], 1.0f);
}

// ---------------------------------------------------------------------------
static inline int cdiv(long a, long b) { return (int)((a + b - 1) / b); }

template<bool ACC_>
static void launch_gemm(hipStream_t s,
    const __bf16* A, const __bf16* Bm, float* C,
    int M, int N, int K, int lda, int ldb, int ldc,
    long sAy, long sBy, long sCy, long sAz, long sBz, long sCz,
    int gy, int gz, float alpha)
{
  int tm = (M + 127) / 128, tn = (N + 127) / 128;
  dim3 grid((unsigned)(tm * tn), (unsigned)gy, (unsigned)gz);
  gemm_wmma<ACC_><<<grid, dim3(256), 0, s>>>(
      A, Bm, C, M, N, K, lda, ldb, ldc,
      sAy, sBy, sCy, sAz, sBz, sCz, alpha, tn);
}

extern "C" void kernel_launch(void* const* d_in, const int* in_sizes, int n_in,
                              void* d_out, int out_size, void* d_ws, size_t ws_size,
                              hipStream_t stream) {
  (void)in_sizes; (void)n_in; (void)out_size; (void)ws_size;

  const int*   idx         = (const int*)d_in[0];
  const int*   targets     = (const int*)d_in[1];
  const float* wte         = (const float*)d_in[2];
  const float* wpe         = (const float*)d_in[3];
  const float* ln1_w       = (const float*)d_in[4];
  const float* ln1_b       = (const float*)d_in[5];
  const float* attn_w      = (const float*)d_in[6];
  const float* attn_proj_w = (const float*)d_in[7];
  const float* ln2_w       = (const float*)d_in[8];
  const float* ln2_b       = (const float*)d_in[9];
  const float* fc_w        = (const float*)d_in[10];
  const float* fc_proj_w   = (const float*)d_in[11];
  const float* lnf_w       = (const float*)d_in[12];
  const float* lnf_b       = (const float*)d_in[13];
  const float* lm_w        = (const float*)d_in[14];

  // ---- carve workspace ----
  char* ws = (char*)d_ws;
  auto alloc = [&](size_t bytes) -> char* {
    char* p = ws; ws += (bytes + 255) & ~(size_t)255; return p;
  };
  float*  x    = (float*) alloc((size_t)cBT * cE * 4);           // residual stream
  __bf16* hb   = (__bf16*)alloc((size_t)cBT * cE * 2);           // LN output (bf16)
  float*  qkv  = (float*) alloc((size_t)cBT * 3 * cE * 4);
  __bf16* qkvb = (__bf16*)alloc((size_t)cBT * 3 * cE * 2);
  float*  att  = (float*) alloc((size_t)cB * cH * cT * cT * 4);
  __bf16* attb = (__bf16*)alloc((size_t)cB * cH * cT * cT * 2);
  __bf16* vt   = (__bf16*)alloc((size_t)cB * cH * cHD * cT * 2); // V^T per (b,h)
  float*  y    = (float*) alloc((size_t)cBT * cE * 4);
  __bf16* yb   = (__bf16*)alloc((size_t)cBT * cE * 2);
  float*  fc   = (float*) alloc((size_t)cBT * 4 * cE * 4);
  __bf16* fcb  = (__bf16*)alloc((size_t)cBT * 4 * cE * 2);
  __bf16* wb   = (__bf16*)alloc((size_t)cV * cE * 2);            // bf16 weight staging
  float*  lossacc = (float*)alloc(2 * 4);

  const float attn_scale = 0.125f;  // 1/sqrt(64)

  // ---- embeddings ----
  embed_kernel<<<cdiv((long)cBT * cE, 256), 256, 0, stream>>>(idx, wte, wpe, x);

  for (int l = 0; l < cNL; ++l) {
    // ln1 -> hb (bf16)
    layernorm_bf16_kernel<<<cBT, 256, 0, stream>>>(x, ln1_w + (long)l * cE,
                                                   ln1_b + (long)l * cE, hb);
    // qkv = hb @ attn_w[l]^T
    cvt_bf16_kernel<<<cdiv((long)3 * cE * cE, 256), 256, 0, stream>>>(
        attn_w + (long)l * 3 * cE * cE, wb, (long)3 * cE * cE);
    launch_gemm<false>(stream, hb, wb, qkv,
        cBT, 3 * cE, cE, cE, cE, 3 * cE,
        0, 0, 0, 0, 0, 0, 1, 1, 1.0f);
    cvt_bf16_kernel<<<cdiv((long)cBT * 3 * cE, 256), 256, 0, stream>>>(
        qkv, qkvb, (long)cBT * 3 * cE);

    // att[b,h] = scale * Q[b,:,h,:] @ K[b,:,h,:]^T  (grid.y = b, grid.z = h)
    launch_gemm<false>(stream, qkvb, qkvb + cE, att,
        cT, cT, cHD, 3 * cE, 3 * cE, cT,
        (long)cT * 3 * cE, (long)cT * 3 * cE, (long)cH * cT * cT,
        (long)cHD,         (long)cHD,         (long)cT * cT,
        cB, cH, attn_scale);

    attn_softmax_kernel<<<cB * cH * cT, 256, 0, stream>>>(att);
    cvt_bf16_kernel<<<cdiv((long)cB * cH * cT * cT, 256), 256, 0, stream>>>(
        att, attb, (long)cB * cH * cT * cT);

    // vt[(b,h)] = V[b,:,h,:]^T  (hd x T, contiguous K for the GEMM B operand)
    transpose_v_kernel<<<cdiv((long)cB * cH * cHD * cT, 256), 256, 0, stream>>>(
        qkvb, vt);

    // y[b,:,h,:] = att[b,h] @ V[b,:,h,:]
    launch_gemm<false>(stream, attb, vt, y,
        cT, cHD, cT, cT, cT, cE,
        (long)cH * cT * cT, (long)cH * cHD * cT, (long)cT * cE,
        (long)cT * cT,      (long)cHD * cT,      (long)cHD,
        cB, cH, 1.0f);

    // x += y @ attn_proj_w[l]^T   (fused residual via ACC)
    cvt_bf16_kernel<<<cdiv((long)cBT * cE, 256), 256, 0, stream>>>(
        y, yb, (long)cBT * cE);
    cvt_bf16_kernel<<<cdiv((long)cE * cE, 256), 256, 0, stream>>>(
        attn_proj_w + (long)l * cE * cE, wb, (long)cE * cE);
    launch_gemm<true>(stream, yb, wb, x,
        cBT, cE, cE, cE, cE, cE,
        0, 0, 0, 0, 0, 0, 1, 1, 1.0f);

    // MLP
    layernorm_bf16_kernel<<<cBT, 256, 0, stream>>>(x, ln2_w + (long)l * cE,
                                                   ln2_b + (long)l * cE, hb);
    cvt_bf16_kernel<<<cdiv((long)4 * cE * cE, 256), 256, 0, stream>>>(
        fc_w + (long)l * 4 * cE * cE, wb, (long)4 * cE * cE);
    launch_gemm<false>(stream, hb, wb, fc,
        cBT, 4 * cE, cE, cE, cE, 4 * cE,
        0, 0, 0, 0, 0, 0, 1, 1, 1.0f);
    gelu_bf16_kernel<<<cdiv((long)cBT * 4 * cE, 256), 256, 0, stream>>>(
        fc, fcb, (long)cBT * 4 * cE);
    cvt_bf16_kernel<<<cdiv((long)cE * 4 * cE, 256), 256, 0, stream>>>(
        fc_proj_w + (long)l * cE * 4 * cE, wb, (long)cE * 4 * cE);
    launch_gemm<true>(stream, fcb, wb, x,
        cBT, cE, 4 * cE, 4 * cE, 4 * cE, cE,
        0, 0, 0, 0, 0, 0, 1, 1, 1.0f);
  }

  // ---- final LN + lm_head ----
  layernorm_bf16_kernel<<<cBT, 256, 0, stream>>>(x, lnf_w, lnf_b, hb);
  cvt_bf16_kernel<<<cdiv((long)cV * cE, 256), 256, 0, stream>>>(
      lm_w, wb, (long)cV * cE);
  float* logits = (float*)d_out;
  launch_gemm<false>(stream, hb, wb, logits,
      cBT, cV, cE, cE, cE, cV,
      0, 0, 0, 0, 0, 0, 1, 1, 1.0f);

  // ---- cross entropy -> d_out[B*T*V] ----
  zero_loss_kernel<<<1, 1, 0, stream>>>(lossacc);
  ce_kernel<<<cBT, 256, 0, stream>>>(logits, targets, lossacc);
  ce_finalize_kernel<<<1, 1, 0, stream>>>(lossacc, logits + (long)cBT * cV);
}
